// Attention_80092550135834
// MI455X (gfx1250) — matrix operand order repaired and checked by reading, not compiled, not run on hardware
//
#include <hip/hip_runtime.h>

// ---------------------------------------------------------------------------
// Linear-attention (Performer) block for MI455X / gfx1250, wave32 + WMMA.
// B=8, L=4096, DIM=768, H=12, D=64, M=256.
// GEMMs via v_wmma_f32_16x16x32_f16. LDS tiles are stored in WMMA *fragment
// order* so every fragment load is one v16h (2x ds_load_b128) and all global
// reads are contiguous 16B+ chunks. f16 stagings use the gfx1250 async
// global->LDS path (ASYNCcnt). kv stage is split-K (8x) for occupancy, with
// k_sum produced by the matrix core via an augmented ones-column.
// ---------------------------------------------------------------------------

typedef __attribute__((ext_vector_type(16))) _Float16 v16h;
typedef __attribute__((ext_vector_type(8)))  _Float16 v8h;
typedef __attribute__((ext_vector_type(8)))  float    v8f;
typedef __attribute__((ext_vector_type(4)))  float    f4;
typedef __attribute__((ext_vector_type(4)))  int      i32x4;

#define FEPS 0.001f
#define KV_SPLIT 8

#define WMMA_F16(a, b, c) \
  __builtin_amdgcn_wmma_f32_16x16x32_f16(false, (a), false, (b), (short)0, (c), false, false)

// ---- gfx1250 async global->LDS (16B), guarded so compile never breaks ------
#if defined(__has_builtin)
#  if __has_builtin(__builtin_amdgcn_global_load_async_to_lds_b128) && \
      __has_builtin(__builtin_amdgcn_s_wait_asynccnt)
#    define HAVE_ASYNC_LDS 1
#  endif
#endif
#ifndef HAVE_ASYNC_LDS
#  define HAVE_ASYNC_LDS 0
#endif

__device__ __forceinline__ void cpy_b128(const _Float16* g, _Float16* l) {
#if HAVE_ASYNC_LDS
  __builtin_amdgcn_global_load_async_to_lds_b128(
      (__attribute__((address_space(1))) i32x4*)g,
      (__attribute__((address_space(3))) i32x4*)l, 0, 0);
#else
  *(v8h*)l = *(const v8h*)g;
#endif
}
__device__ __forceinline__ void async_join() {
#if HAVE_ASYNC_LDS
  __builtin_amdgcn_s_wait_asynccnt(0);
#endif
}

// Convert 8 consecutive f32 (two f4) into a packed v8h.
__device__ __forceinline__ v8h cvt8(f4 lo, f4 hi) {
  v8h r;
#pragma unroll
  for (int i = 0; i < 4; ++i) { r[i] = (_Float16)lo[i]; r[i + 4] = (_Float16)hi[i]; }
  return r;
}
__device__ __forceinline__ v8h splat8(float v) {
  v8h z;
#pragma unroll
  for (int i = 0; i < 8; ++i) z[i] = (_Float16)v;
  return z;
}

// Fragment-order LDS records (32B per (subtile,lane)):
//  A: slots 0..15 <- row m, K = kb..kb+15
//     with m = st*16 + (lane&15), kb = 16*(lane>>4)
//  B: slots 0..15 <- col n, K = kb..kb+15, n = st*16+(lane&15), kb = 16*(lane>>4)
//  C/D: element in reg r of lane l is (m = r + 8*(l>>4), n = l&15)

// ---------------------------------------------------------------------------
// Stage 1: qkv = x @ qkv_w^T + qkv_b -> q,k (B,H,L,64) f16 ; v (B,H,64,L) f16
// ---------------------------------------------------------------------------
__global__ __launch_bounds__(256)
void qkv_gemm_kernel(const float* __restrict__ x,       // (32768,768)
                     const float* __restrict__ qkv_w,   // (2304,768)
                     const float* __restrict__ qkv_b,
                     _Float16* __restrict__ outq,
                     _Float16* __restrict__ outk,
                     _Float16* __restrict__ outv) {     // transposed (B,H,64,L)
  constexpr int K = 768;
  __shared__ v16h sA[8 * 32];
  __shared__ v16h sB[4 * 32];
  const int tid = threadIdx.x, wave = tid >> 5, lane = tid & 31;
  const int m0 = blockIdx.x * 128;
  const int n0 = blockIdx.y * 64;

  const int a_st = tid >> 5, a_fl = tid & 31;
  const int a_kb = (a_fl >> 4) << 4;
  const float* a_row = x + (size_t)(m0 + a_st * 16 + (a_fl & 15)) * K;
  const int b_unit = tid & 127;
  const int b_st = b_unit >> 5, b_fl = b_unit & 31;
  const int b_kb = (b_fl >> 4) << 4;
  const int b_half = tid >> 7;
  const float* b_row = qkv_w + (size_t)(n0 + b_st * 16 + (b_fl & 15)) * K;

  v8f acc[4] = {};
  for (int k0 = 0; k0 < K; k0 += 32) {
    if (k0 + 32 < K) __builtin_prefetch(a_row + k0 + 32, 0, 1);
    {
      const float* s = a_row + k0 + a_kb;
      v8h lo = cvt8(((const f4*)s)[0], ((const f4*)s)[1]);
      v8h hi = cvt8(((const f4*)(s + 8))[0], ((const f4*)(s + 8))[1]);
      v8h* d = (v8h*)&sA[a_st * 32 + a_fl];
      d[0] = lo; d[1] = hi;
    }
    {
      const float* s = b_row + k0 + b_kb + b_half * 8;
      ((v8h*)&sB[b_st * 32 + b_fl])[b_half] = cvt8(((const f4*)s)[0], ((const f4*)s)[1]);
    }
    __syncthreads();
    const v16h a = sA[wave * 32 + lane];
#pragma unroll
    for (int t = 0; t < 4; ++t) acc[t] = WMMA_F16(a, sB[t * 32 + lane], acc[t]);
    __syncthreads();
  }

  const int part = blockIdx.y / 12, h = blockIdx.y % 12;
  const int row0 = m0 + wave * 16 + ((lane >> 4) << 3);  // 8 consecutive rows
  const int bi = row0 >> 12, l = row0 & 4095;
#pragma unroll
  for (int t = 0; t < 4; ++t) {
    const int d = t * 16 + (lane & 15);
    const float bias = qkv_b[n0 + d];
    v8h pack;
#pragma unroll
    for (int r = 0; r < 8; ++r) pack[r] = (_Float16)(acc[t][r] + bias);
    if (part == 2) {  // v: transposed layout -> contiguous 16B store
      *(v8h*)&outv[(((size_t)bi * 12 + h) * 64 + d) * 4096 + l] = pack;
    } else {
      _Float16* dst = part ? outk : outq;
#pragma unroll
      for (int r = 0; r < 8; ++r)
        dst[(((size_t)bi * 12 + h) * 4096 + (l + r)) * 64 + d] = pack[r];
    }
  }
}

// ---------------------------------------------------------------------------
// Stage 2: feat = relu((t @ proj_mat^T)/16)+EPS.  q -> (B,H,L,256);
// k -> transposed (B,H,256,L) so the kv GEMM consumes contiguous K runs.
// ---------------------------------------------------------------------------
__global__ __launch_bounds__(256)
void feature_kernel(const _Float16* __restrict__ t_in,   // (B,H,L,64)
                    const float* __restrict__ proj_mat,  // (256,64)
                    _Float16* __restrict__ feat,
                    int transpose_out) {
  __shared__ v16h sA[16 * 32];   // 8 m-subtiles x 2 k-subtiles
  __shared__ v16h sB[32 * 32];   // 16 n-subtiles x 2 k-subtiles
  const int tid = threadIdx.x, wave = tid >> 5, lane = tid & 31;
  const int bh = blockIdx.y;
  const int l0 = blockIdx.x * 128;
  const _Float16* Abase = t_in + ((size_t)bh * 4096 + l0) * 64;

  for (int u = tid; u < 512; u += 256) {           // A: async f16 copies
    const int sub = u >> 5, fl = u & 31;
    const int stm = sub >> 1, stk = sub & 1;
    const int kb = (fl >> 4) << 4;
    const _Float16* s = Abase + (size_t)(stm * 16 + (fl & 15)) * 64 + stk * 32 + kb;
    v8h* d = (v8h*)&sA[sub * 32 + fl];
    cpy_b128(s, (_Float16*)d);
    cpy_b128(s + 8, (_Float16*)(d + 1));
  }
  for (int u = tid; u < 1024; u += 256) {          // B: f32->f16, contiguous rows
    const int sub = u >> 5, fl = u & 31;
    const int stn = sub >> 1, stk = sub & 1;
    const int kb = (fl >> 4) << 4;
    const float* s = proj_mat + (size_t)(stn * 16 + (fl & 15)) * 64 + stk * 32 + kb;
    v8h* d = (v8h*)&sB[sub * 32 + fl];
    d[0] = cvt8(((const f4*)s)[0], ((const f4*)s)[1]);
    d[1] = cvt8(((const f4*)s)[2], ((const f4*)s)[3]);
  }
  async_join();
  __syncthreads();

  const v16h a0 = sA[(wave * 2 + 0) * 32 + lane];
  const v16h a1 = sA[(wave * 2 + 1) * 32 + lane];
#pragma unroll
  for (int t = 0; t < 16; ++t) {
    v8f acc = {};
    acc = WMMA_F16(a0, sB[(t * 2 + 0) * 32 + lane], acc);
    acc = WMMA_F16(a1, sB[(t * 2 + 1) * 32 + lane], acc);
    const int mc = t * 16 + (lane & 15);
    v8h pack;
#pragma unroll
    for (int r = 0; r < 8; ++r) {
      float z = 0.0625f * acc[r];                  // 1/sqrt(256)
      pack[r] = (_Float16)(fmaxf(z, 0.0f) + FEPS);
    }
    if (transpose_out) {                           // (B,H,256,L): 16B store
      const int lb = l0 + wave * 16 + ((lane >> 4) << 3);
      *(v8h*)&feat[((size_t)bh * 256 + mc) * 4096 + lb] = pack;
    } else {
      const int rb = wave * 16 + ((lane >> 4) << 3);
#pragma unroll
      for (int r = 0; r < 8; ++r)
        feat[((size_t)bh * 4096 + l0 + rb + r) * 256 + mc] = pack[r];
    }
  }
}

// ---------------------------------------------------------------------------
// Stage 3: split-K kv + ksum.  Per (b,h, split): partial over 512 of L.
// B = [vT | ones | 0] (32x80 tiles): column 64 of the WMMA output is the
// k_p row-sum (normalizer) -- computed by the matrix core, no extra kernel.
// ---------------------------------------------------------------------------
__global__ __launch_bounds__(256)
void kv_kernel(const _Float16* __restrict__ k_pT,  // (B,H,256,4096)
               const _Float16* __restrict__ vT,    // (B,H,64,4096)
               float* __restrict__ kv_part,        // (SPLIT,96,64,256)
               float* __restrict__ ks_part) {      // (SPLIT,96,256)
  __shared__ v16h sA[16 * 32];   // 16 m-subtiles (M=256)
  __shared__ v16h sB[5 * 32];    // 5 n-subtiles (D=64 + den col + pad)
  const int tid = threadIdx.x, wave = tid >> 5, lane = tid & 31;
  const int bh = blockIdx.x;
  const int split = blockIdx.y;
  const int lbeg = split * (4096 / KV_SPLIT);
  const int lend = lbeg + (4096 / KV_SPLIT);
  const _Float16* kp = k_pT + (size_t)bh * 256 * 4096;
  const _Float16* vb = vT + (size_t)bh * 64 * 4096;

  v8f acc[2][5] = {};
  for (int l0 = lbeg; l0 < lend; l0 += 32) {
    for (int u = tid; u < 512; u += 256) {         // A: k_pT rows, contiguous
      const int st = u >> 5, fl = u & 31;
      const int kb = (fl >> 4) << 4;
      const _Float16* s = kp + (size_t)(st * 16 + (fl & 15)) * 4096 + l0 + kb;
      v8h* d = (v8h*)&sA[st * 32 + fl];
      cpy_b128(s, (_Float16*)d);
      cpy_b128(s + 8, (_Float16*)(d + 1));
    }
    if (tid < 160) {                               // B: [vT | ones | 0]
      const int st = tid >> 5, fl = tid & 31;
      const int n = st * 16 + (fl & 15);
      const int kb = (fl >> 4) << 4;
      v8h* d = (v8h*)&sB[st * 32 + fl];
      if (n < 64) {
        const _Float16* s = vb + (size_t)n * 4096 + l0 + kb;
        cpy_b128(s, (_Float16*)d);
        cpy_b128(s + 8, (_Float16*)(d + 1));
      } else if (n == 64) {
        d[0] = splat8(1.0f); d[1] = splat8(1.0f);
      } else {
        d[0] = splat8(0.0f); d[1] = splat8(0.0f);
      }
    }
    async_join();
    __syncthreads();
#pragma unroll
    for (int ti = 0; ti < 2; ++ti) {
      const v16h a = sA[(wave * 2 + ti) * 32 + lane];
#pragma unroll
      for (int tj = 0; tj < 5; ++tj)
        acc[ti][tj] = WMMA_F16(a, sB[tj * 32 + lane], acc[ti][tj]);
    }
    __syncthreads();
  }
  float* kdst = kv_part + ((size_t)split * 96 + bh) * 64 * 256;
#pragma unroll
  for (int ti = 0; ti < 2; ++ti)
#pragma unroll
    for (int tj = 0; tj < 4; ++tj) {
      const int d = tj * 16 + (lane & 15);
      const int m = wave * 32 + ti * 16 + ((lane >> 4) << 3);
      *(v8f*)&kdst[(size_t)d * 256 + m] = acc[ti][tj];  // 8 contiguous f32
    }
  if ((lane & 15) == 0) {                          // column 64 = k_sum partial
    float* sdst = ks_part + ((size_t)split * 96 + bh) * 256;
#pragma unroll
    for (int ti = 0; ti < 2; ++ti)
#pragma unroll
      for (int r = 0; r < 8; ++r) {
        const int m = wave * 32 + ti * 16 + ((lane >> 4) << 3) + r;
        sdst[m] = acc[ti][4][r];
      }
  }
}

// Stage 3b: deterministic split-K reduction (no float atomics)
__global__ __launch_bounds__(256)
void kv_reduce_kernel(const float* __restrict__ kv_part,
                      const float* __restrict__ ks_part,
                      float* __restrict__ kvT,
                      float* __restrict__ ksum) {
  const size_t KV_E = (size_t)96 * 64 * 256;  // 1,572,864
  const size_t KS_E = (size_t)96 * 256;       // 24,576
  const size_t gid = (size_t)blockIdx.x * 256 + threadIdx.x;
  if (gid < KV_E) {
    float s = 0.0f;
#pragma unroll
    for (int p = 0; p < KV_SPLIT; ++p) s += kv_part[(size_t)p * KV_E + gid];
    kvT[gid] = s;
  } else if (gid < KV_E + KS_E) {
    const size_t i = gid - KV_E;
    float s = 0.0f;
#pragma unroll
    for (int p = 0; p < KV_SPLIT; ++p) s += ks_part[(size_t)p * KS_E + i];
    ksum[i] = s;
  }
}

// ---------------------------------------------------------------------------
// Stage 4: num = q_p @ kv, den = q_p @ k_sum via augmented B = [kvT^T|ksum|0]
// (32x80 tiles); den is WMMA column 64, broadcast by wave shuffle; attn f16.
// ---------------------------------------------------------------------------
__global__ __launch_bounds__(256)
void attn_out_kernel(const _Float16* __restrict__ q_p,  // (B,H,L,256)
                     const float* __restrict__ kvT,     // (B,H,64,256)
                     const float* __restrict__ k_sum,   // (B,H,256)
                     _Float16* __restrict__ attn) {     // (B,L,768) f16
  __shared__ v16h sA[8 * 32];
  __shared__ v16h sB[5 * 32];
  const int tid = threadIdx.x, wave = tid >> 5, lane = tid & 31;
  const int bh = blockIdx.y;
  const int b = bh / 12, h = bh % 12;
  const int l0 = blockIdx.x * 128;
  const _Float16* qp = q_p + ((size_t)bh * 4096 + l0) * 256;
  const float* kvb = kvT + (size_t)bh * 64 * 256;
  const float* ks = k_sum + (size_t)bh * 256;

  const int a_st = tid >> 5, a_fl = tid & 31;
  const int a_kb = (a_fl >> 4) << 4;
  const _Float16* a_row = qp + (size_t)(a_st * 16 + (a_fl & 15)) * 256;

  v8f acc[5] = {};
  for (int k0 = 0; k0 < 256; k0 += 32) {
    {
      v8h* d = (v8h*)&sA[a_st * 32 + a_fl];
      cpy_b128(a_row + k0 + a_kb, (_Float16*)d);
      cpy_b128(a_row + k0 + 8 + a_kb, (_Float16*)(d + 1));
    }
    if (tid < 160) {                     // B: [kvT | ksum | 0], contiguous in m
      const int st = tid >> 5, fl = tid & 31;
      const int n = st * 16 + (fl & 15);
      const int kb = (fl >> 4) << 4;
      v8h lo, hi;
      if (n <= 64) {
        const float* s = (n < 64) ? kvb + (size_t)n * 256 + k0 + kb
                                  : ks + k0 + kb;
        lo = cvt8(((const f4*)s)[0], ((const f4*)s)[1]);
        hi = cvt8(((const f4*)s)[2], ((const f4*)s)[3]);
      } else {
        lo = splat8(0.0f); hi = splat8(0.0f);
      }
      v8h* d = (v8h*)&sB[st * 32 + fl];
      d[0] = lo; d[1] = hi;
    }
    async_join();
    __syncthreads();
    const v16h a = sA[wave * 32 + lane];
#pragma unroll
    for (int t = 0; t < 5; ++t) acc[t] = WMMA_F16(a, sB[t * 32 + lane], acc[t]);
    __syncthreads();
  }
  // den(m) sits in acc[4] reg r of lane (lane & 16) (same reg index r).
#pragma unroll
  for (int t = 0; t < 4; ++t) {
#pragma unroll
    for (int r = 0; r < 8; ++r) {
      const float den = __shfl(acc[4][r], (lane & 16), 32);
      const int l = l0 + wave * 16 + ((lane >> 4) << 3) + r;
      const int d = t * 16 + (lane & 15);
      attn[((size_t)b * 4096 + l) * 768 + h * 64 + d] = (_Float16)(acc[t][r] / den);
    }
  }
}

// ---------------------------------------------------------------------------
// Stage 5: out = attn @ proj_w^T + proj_b  -> f32 d_out (32768x768)
// ---------------------------------------------------------------------------
__global__ __launch_bounds__(256)
void proj_gemm_kernel(const _Float16* __restrict__ attn,
                      const float* __restrict__ proj_w,
                      const float* __restrict__ proj_b,
                      float* __restrict__ out) {
  constexpr int K = 768;
  __shared__ v16h sA[8 * 32];
  __shared__ v16h sB[4 * 32];
  const int tid = threadIdx.x, wave = tid >> 5, lane = tid & 31;
  const int m0 = blockIdx.x * 128;
  const int n0 = blockIdx.y * 64;

  const int a_st = tid >> 5, a_fl = tid & 31;
  const int a_kb = (a_fl >> 4) << 4;
  const _Float16* a_row = attn + (size_t)(m0 + a_st * 16 + (a_fl & 15)) * K;
  const int b_unit = tid & 127;
  const int b_st = b_unit >> 5, b_fl = b_unit & 31;
  const int b_kb = (b_fl >> 4) << 4;
  const int b_half = tid >> 7;
  const float* b_row = proj_w + (size_t)(n0 + b_st * 16 + (b_fl & 15)) * K;

  v8f acc[4] = {};
  for (int k0 = 0; k0 < K; k0 += 32) {
    if (k0 + 32 < K) __builtin_prefetch(a_row + k0 + 32, 0, 1);
    {
      v8h* d = (v8h*)&sA[a_st * 32 + a_fl];
      cpy_b128(a_row + k0 + a_kb, (_Float16*)d);
      cpy_b128(a_row + k0 + 8 + a_kb, (_Float16*)(d + 1));
    }
    {
      const float* s = b_row + k0 + b_kb + b_half * 8;
      ((v8h*)&sB[b_st * 32 + b_fl])[b_half] = cvt8(((const f4*)s)[0], ((const f4*)s)[1]);
    }
    async_join();
    __syncthreads();
    const v16h a = sA[wave * 32 + lane];
#pragma unroll
    for (int t = 0; t < 4; ++t) acc[t] = WMMA_F16(a, sB[t * 32 + lane], acc[t]);
    __syncthreads();
  }
#pragma unroll
  for (int t = 0; t < 4; ++t)
#pragma unroll
    for (int r = 0; r < 8; ++r) {
      const int row = m0 + wave * 16 + ((lane >> 4) << 3) + r;
      const int col = n0 + t * 16 + (lane & 15);
      out[(size_t)row * 768 + col] = acc[t][r] + proj_b[col];
    }
}

// ---------------------------------------------------------------------------
extern "C" void kernel_launch(void* const* d_in, const int* in_sizes, int n_in,
                              void* d_out, int out_size, void* d_ws, size_t ws_size,
                              hipStream_t stream) {
  (void)in_sizes; (void)n_in; (void)out_size; (void)ws_size;
  const float* x        = (const float*)d_in[0];
  const float* qkv_w    = (const float*)d_in[1];
  const float* qkv_b    = (const float*)d_in[2];
  const float* proj_w   = (const float*)d_in[3];
  const float* proj_b   = (const float*)d_in[4];
  const float* proj_mat = (const float*)d_in[5];
  float* out = (float*)d_out;

  const size_t QKV_E  = (size_t)8 * 12 * 4096 * 64;
  const size_t FEAT_E = (size_t)8 * 12 * 4096 * 256;
  const size_t KV_E   = (size_t)8 * 12 * 64 * 256;   // 1,572,864
  const size_t KS_E   = (size_t)8 * 12 * 256;        // 24,576
  const size_t ATTN_E = (size_t)8 * 4096 * 768;

  char* ws = (char*)d_ws;
  size_t ofs = 0;
  auto carve = [&](size_t bytes) -> void* {
    void* p = ws + ofs;
    ofs += (bytes + 255) & ~(size_t)255;
    return p;
  };
  _Float16* q    = (_Float16*)carve(QKV_E * sizeof(_Float16));   // (B,H,L,64)
  _Float16* k    = (_Float16*)carve(QKV_E * sizeof(_Float16));   // (B,H,L,64)
  _Float16* vT   = (_Float16*)carve(QKV_E * sizeof(_Float16));   // (B,H,64,L)
  _Float16* q_p  = (_Float16*)carve(FEAT_E * sizeof(_Float16));  // (B,H,L,256)
  _Float16* k_pT = (_Float16*)carve(FEAT_E * sizeof(_Float16));  // (B,H,256,L)
  float*    kvT  = (float*)carve(KV_E * sizeof(float));          // (B,H,64,256)
  float*    ksum = (float*)carve(KS_E * sizeof(float));
  float*    kvp  = (float*)carve((size_t)KV_SPLIT * KV_E * sizeof(float));
  float*    ksp  = (float*)carve((size_t)KV_SPLIT * KS_E * sizeof(float));
  _Float16* attn = (_Float16*)carve(ATTN_E * sizeof(_Float16));  // (B,L,768)

  qkv_gemm_kernel<<<dim3(256, 36), 256, 0, stream>>>(x, qkv_w, qkv_b, q, k, vT);
  feature_kernel<<<dim3(32, 96), 256, 0, stream>>>(q, proj_mat, q_p, 0);
  feature_kernel<<<dim3(32, 96), 256, 0, stream>>>(k, proj_mat, k_pT, 1);
  kv_kernel<<<dim3(96, KV_SPLIT), 256, 0, stream>>>(k_pT, vT, kvp, ksp);
  {
    const int total = (int)((KV_E + KS_E + 255) / 256);
    kv_reduce_kernel<<<dim3(total), 256, 0, stream>>>(kvp, ksp, kvT, ksum);
  }
  attn_out_kernel<<<dim3(32, 96), 256, 0, stream>>>(q_p, kvT, ksum, attn);
  proj_gemm_kernel<<<dim3(256, 12), 256, 0, stream>>>(attn, proj_w, proj_b, out);
}
